// GNN_2121713844788
// MI455X (gfx1250) — matrix operand order repaired and checked by reading, not compiled, not run on hardware
//
#include <hip/hip_runtime.h>

// GCN 2-layer: N=100000 nodes (multiple of 16), E=1.6M edges, 1 -> 128 -> 128.
#define FDIM 128

typedef __attribute__((ext_vector_type(16))) __bf16 v16bf;
typedef __attribute__((ext_vector_type(4)))  __bf16 v4bf;
typedef __attribute__((ext_vector_type(8)))  float  v8f;
typedef __attribute__((ext_vector_type(4)))  float  v4f;

// ---- degree: deg[i] = 1 (self loop) + #(dst == i) -------------------------
__global__ __launch_bounds__(256) void k_init_deg(float* __restrict__ deg, int n) {
    int i = blockIdx.x * 256 + threadIdx.x;
    if (i < n) deg[i] = 1.0f;
}

__global__ __launch_bounds__(256) void k_count_deg(const int* __restrict__ dst,
                                                   float* __restrict__ deg, int E) {
    int e = blockIdx.x * 256 + threadIdx.x;
    if (e < E) atomicAdd(&deg[dst[e]], 1.0f);
}

// dinv = rsqrt(deg) (in place); xs = x*dinv; s init with self-loop term xs[i]
__global__ __launch_bounds__(256) void k_dinv_xs(const float* __restrict__ x,
                                                 float* __restrict__ deg_dinv,
                                                 float* __restrict__ s,
                                                 float* __restrict__ xs, int n) {
    int i = blockIdx.x * 256 + threadIdx.x;
    if (i >= n) return;
    float d = rsqrtf(deg_dinv[i]);
    deg_dinv[i] = d;
    float v = x[i] * d;
    xs[i] = v;
    s[i]  = v;
}

// layer-1 scalar aggregation: s[dst] += xs[src]
__global__ __launch_bounds__(256) void k_scatter_s(const int* __restrict__ src,
                                                   const int* __restrict__ dst,
                                                   const float* __restrict__ xs,
                                                   float* __restrict__ s, int E) {
    int e = blockIdx.x * 256 + threadIdx.x;
    if (e < E) atomicAdd(&s[dst[e]], xs[src[e]]);
}

// h[i,j] = relu(dinv[i]*s[i]*W1[j] + b1[j]);  g = h*dinv (bf16);
// acc (= d_out, f32) initialized with the layer-2 self-loop message g[i,:].
__global__ __launch_bounds__(256) void k_build(const float* __restrict__ s,
                                               const float* __restrict__ dinv,
                                               const float* __restrict__ W1,
                                               const float* __restrict__ b1,
                                               float* __restrict__ acc,
                                               __bf16* __restrict__ g, int n) {
    int t = blockIdx.x * 256 + threadIdx.x;
    if (t >= n * FDIM) return;
    int i = t >> 7;
    int j = t & (FDIM - 1);
    float di  = dinv[i];
    float agg = s[i] * di;
    float h   = fmaxf(fmaf(agg, W1[j], b1[j]), 0.0f);
    float gf  = h * di;
    acc[t] = gf;             // self-loop message, f32 accumulator
    g[t]   = (__bf16)gf;     // bf16 copy for edge gathers
}

// W2^T in bf16: w2t[n*128+k] = W2[k*128+n]  (B operand becomes contiguous)
__global__ __launch_bounds__(256) void k_w2t(const float* __restrict__ W2,
                                             __bf16* __restrict__ w2t) {
    int t = blockIdx.x * 256 + threadIdx.x;
    if (t >= FDIM * FDIM) return;
    int nn = t >> 7, k = t & (FDIM - 1);
    w2t[(nn << 7) + k] = (__bf16)W2[(k << 7) + nn];
}

// layer-2 vector aggregation: acc[dst,:] += g[src,:]  (32 lanes/edge, 4 feats/lane)
__global__ __launch_bounds__(256) void k_scatter_vec(const int* __restrict__ src,
                                                     const int* __restrict__ dst,
                                                     const __bf16* __restrict__ g,
                                                     float* __restrict__ acc, int E) {
    int t = blockIdx.x * 256 + threadIdx.x;
    int e = t >> 5;
    if (e >= E) return;
    int lane = t & 31;
    int si = src[e], di = dst[e];
    v4bf gv = *(const v4bf*)(g + (si << 7) + (lane << 2));
    float* p = acc + (di << 7) + (lane << 2);
    atomicAdd(p + 0, (float)gv[0]);
    atomicAdd(p + 1, (float)gv[1]);
    atomicAdd(p + 2, (float)gv[2]);
    atomicAdd(p + 3, (float)gv[3]);
}

// out[i,:] = (dinv[i]*acc[i,:]) @ W2 + b2, in place on acc(=d_out), via WMMA.
// One wave per 16-row tile: A(16x128 bf16, built in registers), 8 N-tiles x 4 K-steps.
__global__ __launch_bounds__(256) void k_gemm(float* __restrict__ acc_out,
                                              const __bf16* __restrict__ w2t,
                                              const float* __restrict__ dinv,
                                              const float* __restrict__ b2,
                                              int ntiles) {
    int wave = threadIdx.x >> 5;
    int rt = blockIdx.x * 8 + wave;
    if (rt >= ntiles) return;               // uniform per wave: EXEC stays all-1s
    int lane  = threadIdx.x & 31;
    int rr    = lane & 15;
    int khalf = lane >> 4;
    int myrow = rt * 16 + rr;
    float dr  = dinv[myrow];

    // A k order chosen here (contiguous, same as B): lane<16 -> K k0..k0+15,
    //                                    lane>=16 -> K k0+16..k0+31
    const float* arow = acc_out + (myrow << 7);
    v16bf a[4];
#pragma unroll
    for (int kt = 0; kt < 4; ++kt) {
        int klo = kt * 32 + khalf * 16;
        v4f lo0 = *(const v4f*)(arow + klo);
        v4f lo1 = *(const v4f*)(arow + klo + 4);
        v4f hi0 = *(const v4f*)(arow + klo + 8);
        v4f hi1 = *(const v4f*)(arow + klo + 12);
#pragma unroll
        for (int q = 0; q < 4; ++q) {
            a[kt][q]      = (__bf16)(lo0[q] * dr);
            a[kt][q + 4]  = (__bf16)(lo1[q] * dr);
            a[kt][q + 8]  = (__bf16)(hi0[q] * dr);
            a[kt][q + 12] = (__bf16)(hi1[q] * dr);
        }
    }

#pragma unroll
    for (int nt = 0; nt < 8; ++nt) {
        int ncol = nt * 16 + rr;
        float bias = b2[ncol];
        v8f c;
#pragma unroll
        for (int r = 0; r < 8; ++r) c[r] = bias;  // fold +b2 into C

        // B k order chosen here (same as A): lane<16 -> K k0..k0+15 of column ncol; lane>=16 -> k0+16..k0+31
        const __bf16* bcol = w2t + (ncol << 7) + khalf * 16;
#pragma unroll
        for (int kt = 0; kt < 4; ++kt) {
            v16bf b = *(const v16bf*)(bcol + kt * 32);
            c = __builtin_amdgcn_wmma_f32_16x16x32_bf16(
                    false, a[kt], false, b, (short)0, c, false, false);
        }
        // D layout: VGPR r -> row rt*16 + khalf*8 + r, col ncol
        float* orow = acc_out + ((rt * 16 + khalf * 8) << 7) + ncol;
#pragma unroll
        for (int r = 0; r < 8; ++r) orow[r * FDIM] = c[r];
    }
}

extern "C" void kernel_launch(void* const* d_in, const int* in_sizes, int n_in,
                              void* d_out, int out_size, void* d_ws, size_t ws_size,
                              hipStream_t stream) {
    const int N = in_sizes[0];          // 100000 (multiple of 16)
    const int E = in_sizes[1] / 2;      // 1600000

    const float* x   = (const float*)d_in[0];
    const int*   src = (const int*)d_in[1];
    const int*   dst = src + E;
    const float* W1  = (const float*)d_in[2];
    const float* b1  = (const float*)d_in[3];
    const float* W2  = (const float*)d_in[4];
    const float* b2  = (const float*)d_in[5];
    float* out = (float*)d_out;

    // workspace layout (~27 MB): dinv | s | xs | g(bf16 N*128) | W2^T(bf16)
    float*  deg = (float*)d_ws;         // becomes dinv in place
    float*  s   = deg + N;
    float*  xs  = s + N;
    __bf16* g   = (__bf16*)(xs + N);
    __bf16* w2t = g + (size_t)N * FDIM;

    const int TB = 256;
    k_init_deg<<<(N + TB - 1) / TB, TB, 0, stream>>>(deg, N);
    k_count_deg<<<(E + TB - 1) / TB, TB, 0, stream>>>(dst, deg, E);
    k_dinv_xs<<<(N + TB - 1) / TB, TB, 0, stream>>>(x, deg, s, xs, N);
    k_scatter_s<<<(E + TB - 1) / TB, TB, 0, stream>>>(src, dst, xs, s, E);
    k_build<<<((long long)N * FDIM + TB - 1) / TB, TB, 0, stream>>>(s, deg, W1, b1, out, g, N);
    k_w2t<<<(FDIM * FDIM + TB - 1) / TB, TB, 0, stream>>>(W2, w2t);
    k_scatter_vec<<<((long long)E * 32 + TB - 1) / TB, TB, 0, stream>>>(src, dst, g, out, E);

    int ntiles = N / 16;                // 6250, exact
    k_gemm<<<(ntiles + 7) / 8, TB, 0, stream>>>(out, w2t, deg, b2, ntiles);
}
